// SchNetEncoderNoEmbed_65678639891016
// MI455X (gfx1250) — matrix operand order repaired and checked by reading, not compiled, not run on hardware
//
#include <hip/hip_runtime.h>
#include <hip/hip_bf16.h>
#include <math.h>

typedef __attribute__((ext_vector_type(16))) _Float16 v16h;
typedef __attribute__((ext_vector_type(4)))  _Float16 v4h;
typedef __attribute__((ext_vector_type(8)))  float    v8f;

#define NODE_DIM 64
#define EDGE_DIM 32
#define HIDDEN_D 128
#define OUT_DIM  32
#define N_LAYERS 3
#define NUM_GRAPHS 128

// ---------------- fragment-order LDS swizzles (ISA 7.12.2 layouts, wave32) ----------------
// A (16x32 f16, MxK): lane = hs*16 + m, hs=(kr>>3)&1, slot = ((kr>>4)<<3)|(kr&7)
// storage: [(ks*32 + lane)*16 + slot]  -> per-lane 16 contiguous halfs (32B aligned)
// 4 consecutive K (not crossing an 8-block) -> 4 consecutive slots (8B aligned when k%4==0)
static __device__ inline int a_swz(int m, int k) {
  int ks = k >> 5, kr = k & 31;
  int hs = (kr >> 3) & 1;
  int slot = ((kr >> 4) << 3) | (kr & 7);
  return ((ks << 5) + (hs << 4) + m) * 16 + slot;
}
// B (32x16 f16, KxN): lane = ((kr>>3)&1)*16 + (n&15), slot = ((kr>>4)<<3)|(kr&7)
// storage: [((ks*NT + nt)*32 + lane)*16 + slot]
// 4 consecutive K (not crossing an 8-block) -> 4 consecutive slots (8B aligned when k%4==0)
static __device__ inline int b_swz(int k, int n, int NT) {
  int ks = k >> 5, kr = k & 31;
  int hs = (kr >> 3) & 1;
  int nt = n >> 4, nl = n & 15;
  return (((ks * NT + nt) << 5) + (hs << 4) + nl) * 16 + (((kr >> 4) << 3) | (kr & 7));
}

static __device__ inline v16h load_frag(const _Float16* p) {
  return *(const v16h*)p;   // 32B contiguous -> 2x ds_load_b128
}

static __device__ inline v8f wmma_f16(v16h a, v16h b, v8f c) {
  return __builtin_amdgcn_wmma_f32_16x16x32_f16(false, a, false, b, (short)0, c, false, false);
}

static __device__ inline v4h to_v4h(float4 v) {
  v4h h;
  h[0] = (_Float16)v.x; h[1] = (_Float16)v.y; h[2] = (_Float16)v.z; h[3] = (_Float16)v.w;
  return h;
}

// ---------------- hd = h @ Wd  (M x 64 @ 64 x 64) ----------------

__global__ __launch_bounds__(128) void dense64_kernel(const float* __restrict__ X,
                                                      const float* __restrict__ W,
                                                      float* __restrict__ Y, int M) {
  __shared__ alignas(32) _Float16 sWf[64 * 64];      // fragment order, NT=4
  __shared__ alignas(32) _Float16 sAf[4][16 * 64];   // per-wave fragment order
  __shared__ alignas(16) float    sOut[4][16 * 64];  // per-wave f32 row-major output tile
  int tid = threadIdx.x, wave = tid >> 5, lane = tid & 31;
  // weights: K blocks of 4 per thread -> one ds_store_b64 each
  for (int j = tid; j < (64 / 4) * 64; j += 128) {
    int kq = j >> 6, n = j & 63;
    int k = kq * 4;
    float4 w = make_float4(W[(k + 0) * 64 + n], W[(k + 1) * 64 + n],
                           W[(k + 2) * 64 + n], W[(k + 3) * 64 + n]);
    *(v4h*)&sWf[b_swz(k, n, 4)] = to_v4h(w);
  }
  int r0 = blockIdx.x * 64 + wave * 16;
  for (int i = lane; i < 16 * 64 / 4; i += 32) {
    int idx = i * 4;
    int r = idx >> 6, c = idx & 63;
    int row = r0 + r;
    float4 v = (row < M) ? *(const float4*)&X[(size_t)row * 64 + c]
                         : make_float4(0.f, 0.f, 0.f, 0.f);
    *(v4h*)&sAf[wave][a_swz(r, c)] = to_v4h(v);
  }
  __syncthreads();
  v8f zero = {};
  v8f acc[4];
#pragma unroll
  for (int nt = 0; nt < 4; ++nt) acc[nt] = zero;
#pragma unroll
  for (int ks = 0; ks < 2; ++ks) {
    v16h a = load_frag(&sAf[wave][((ks << 5) + lane) * 16]);
#pragma unroll
    for (int nt = 0; nt < 4; ++nt) {
      v16h b = load_frag(&sWf[(((ks * 4 + nt) << 5) + lane) * 16]);
      acc[nt] = wmma_f16(a, b, acc[nt]);
    }
  }
  int n = lane & 15, hs = lane >> 4;
#pragma unroll
  for (int nt = 0; nt < 4; ++nt)
#pragma unroll
    for (int j = 0; j < 8; ++j)
      sOut[wave][(j + 8 * hs) * 64 + nt * 16 + n] = acc[nt][j];
  // cooperative coalesced writeback (same-wave DS is in-order; no barrier needed)
  for (int i = lane; i < 16 * 64 / 4; i += 32) {
    int idx = i * 4;
    int r = idx >> 6, c = idx & 63;
    int row = r0 + r;
    if (row < M) *(float4*)&Y[(size_t)row * 64 + c] = *(const float4*)&sOut[wave][idx];
  }
}

// ---------------- fused 2-layer MLP: Y = tanh(X@W1+b1)@W2+b2 (+residual) ----------------

template <int K1, int HID, int K2, bool RES>
__global__ __launch_bounds__(128) void mlp2_kernel(const float* __restrict__ X,
                                                   const float* __restrict__ W1,
                                                   const float* __restrict__ B1,
                                                   const float* __restrict__ W2,
                                                   const float* __restrict__ B2,
                                                   const float* __restrict__ Rin,
                                                   float* __restrict__ Y, int M) {
  constexpr int NT1 = HID / 16, KS1 = K1 / 32;
  constexpr int NT2 = K2 / 16, KS2 = HID / 32;
  static_assert(16 * K2 * 4 <= 16 * HID * 2, "f32 out tile must fit in sHf wave region");
  __shared__ alignas(32) _Float16 sW1f[K1 * HID];
  __shared__ alignas(32) _Float16 sW2f[HID * K2];
  __shared__ float    sB1[HID];
  __shared__ float    sB2[K2];
  __shared__ alignas(32) _Float16 sAf[4][16 * K1];
  __shared__ alignas(32) _Float16 sHf[4][16 * HID];
  int tid = threadIdx.x, wave = tid >> 5, lane = tid & 31;
  for (int j = tid; j < (K1 / 4) * HID; j += 128) {
    int kq = j / HID, n = j % HID;
    int k = kq * 4;
    float4 w = make_float4(W1[(k + 0) * HID + n], W1[(k + 1) * HID + n],
                           W1[(k + 2) * HID + n], W1[(k + 3) * HID + n]);
    *(v4h*)&sW1f[b_swz(k, n, NT1)] = to_v4h(w);
  }
  for (int j = tid; j < (HID / 4) * K2; j += 128) {
    int kq = j / K2, n = j % K2;
    int k = kq * 4;
    float4 w = make_float4(W2[(k + 0) * K2 + n], W2[(k + 1) * K2 + n],
                           W2[(k + 2) * K2 + n], W2[(k + 3) * K2 + n]);
    *(v4h*)&sW2f[b_swz(k, n, NT2)] = to_v4h(w);
  }
  for (int i = tid; i < HID; i += 128) sB1[i] = B1[i];
  for (int i = tid; i < K2; i += 128) sB2[i] = B2[i];
  int r0 = blockIdx.x * 64 + wave * 16;
  for (int i = lane; i < 16 * K1 / 4; i += 32) {
    int idx = i * 4;
    int r = idx / K1, c = idx % K1;
    int row = r0 + r;
    float4 v = (row < M) ? *(const float4*)&X[(size_t)row * K1 + c]
                         : make_float4(0.f, 0.f, 0.f, 0.f);
    *(v4h*)&sAf[wave][a_swz(r, c)] = to_v4h(v);
  }
  __syncthreads();

  int n = lane & 15, hs = lane >> 4;
  v8f zero = {};
  {
    v8f acc[NT1];
#pragma unroll
    for (int nt = 0; nt < NT1; ++nt) acc[nt] = zero;
#pragma unroll
    for (int ks = 0; ks < KS1; ++ks) {
      v16h a = load_frag(&sAf[wave][((ks << 5) + lane) * 16]);
#pragma unroll
      for (int nt = 0; nt < NT1; ++nt) {
        v16h b = load_frag(&sW1f[(((ks * NT1 + nt) << 5) + lane) * 16]);
        acc[nt] = wmma_f16(a, b, acc[nt]);
      }
    }
    // tanh + bias, write intermediate in A-fragment order for GEMM2
#pragma unroll
    for (int nt = 0; nt < NT1; ++nt)
#pragma unroll
      for (int j = 0; j < 8; ++j) {
        int row = j + 8 * hs;
        int col = nt * 16 + n;
        sHf[wave][a_swz(row, col)] = (_Float16)tanhf(acc[nt][j] + sB1[col]);
      }
  }
  __syncthreads();

  v8f acc2[NT2];
#pragma unroll
  for (int nt = 0; nt < NT2; ++nt) acc2[nt] = zero;
#pragma unroll
  for (int ks = 0; ks < KS2; ++ks) {
    v16h a = load_frag(&sHf[wave][((ks << 5) + lane) * 16]);
#pragma unroll
    for (int nt = 0; nt < NT2; ++nt) {
      v16h b = load_frag(&sW2f[(((ks * NT2 + nt) << 5) + lane) * 16]);
      acc2[nt] = wmma_f16(a, b, acc2[nt]);
    }
  }
  // stage f32 output tile in this wave's (now fully consumed) sHf region,
  // then do coalesced float4 writeback; same-wave DS ops are in-order.
  float* sOut = (float*)&sHf[wave][0];
#pragma unroll
  for (int nt = 0; nt < NT2; ++nt)
#pragma unroll
    for (int j = 0; j < 8; ++j)
      sOut[(j + 8 * hs) * K2 + nt * 16 + n] = acc2[nt][j] + sB2[nt * 16 + n];
  for (int i = lane; i < 16 * K2 / 4; i += 32) {
    int idx = i * 4;
    int r = idx / K2, c = idx % K2;
    int row = r0 + r;
    if (row < M) {
      float4 v = *(const float4*)&sOut[idx];
      if (RES) {
        float4 rv = *(const float4*)&Rin[(size_t)row * K2 + c];
        v.x += rv.x; v.y += rv.y; v.z += rv.z; v.w += rv.w;
      }
      *(float4*)&Y[(size_t)row * K2 + c] = v;
    }
  }
}

// ---------------- graph scatter/softmax helpers ----------------

static __device__ inline unsigned fkey(float f) {
  unsigned u = __float_as_uint(f);
  return (u & 0x80000000u) ? ~u : (u | 0x80000000u);
}
static __device__ inline float fdec(unsigned k) {
  unsigned u = (k & 0x80000000u) ? (k & 0x7fffffffu) : ~k;
  return __uint_as_float(u);
}

__global__ void fill_u32_kernel(unsigned* p, unsigned v, int n) {
  int i = blockIdx.x * blockDim.x + threadIdx.x;
  if (i < n) p[i] = v;
}

// scores[e] = sum_d hd[src[e]][d] * ew[e][d] * attn_v[d]   (wave per edge)
__global__ void score_kernel(const float* __restrict__ hd, const float* __restrict__ ew,
                             const int* __restrict__ src, const float* __restrict__ av,
                             float* __restrict__ scores, int E) {
  int e = blockIdx.x * 8 + (threadIdx.x >> 5);
  int lane = threadIdx.x & 31;
  if (e >= E) return;
  int s = src[e];
  float2 a = ((const float2*)(hd + (size_t)s * 64))[lane];
  float2 w = ((const float2*)(ew + (size_t)e * 64))[lane];
  float2 v = ((const float2*)av)[lane];
  float acc = a.x * w.x * v.x + a.y * w.y * v.y;
#pragma unroll
  for (int off = 16; off > 0; off >>= 1) acc += __shfl_xor(acc, off, 32);
  if (lane == 0) scores[e] = acc;
}

__global__ void segmax_kernel(const float* scores, const int* tgt, unsigned* segmax, int E) {
  int e = blockIdx.x * blockDim.x + threadIdx.x;
  if (e < E) atomicMax(&segmax[tgt[e]], fkey(scores[e]));
}

__global__ void expdenom_kernel(float* scores, const int* tgt, const unsigned* segmax,
                                float* denom, int E) {
  int e = blockIdx.x * blockDim.x + threadIdx.x;
  if (e >= E) return;
  int t = tgt[e];
  float es = __expf(scores[e] - fdec(segmax[t]));
  scores[e] = es;  // scores now holds exp_s
  atomicAdd(&denom[t], es);
}

// agg[tgt] += (hd[src]*ew[e]) * (exp_s[e]/denom[tgt])   (wave per edge)
__global__ void scatter_kernel(const float* __restrict__ hd, const float* __restrict__ ew,
                               const float* __restrict__ exps, const float* __restrict__ denom,
                               const int* __restrict__ src, const int* __restrict__ tgt,
                               float* __restrict__ agg, int E) {
  int e = blockIdx.x * 8 + (threadIdx.x >> 5);
  int lane = threadIdx.x & 31;
  if (e >= E) return;
  int s = src[e], t = tgt[e];
  float attn = exps[e] / denom[t];
  float2 a = ((const float2*)(hd + (size_t)s * 64))[lane];
  float2 w = ((const float2*)(ew + (size_t)e * 64))[lane];
  atomicAdd(&agg[(size_t)t * 64 + lane * 2],     a.x * w.x * attn);
  atomicAdd(&agg[(size_t)t * 64 + lane * 2 + 1], a.y * w.y * attn);
}

__global__ void pool_accum_kernel(const float* __restrict__ h, const int* __restrict__ batch,
                                  float* __restrict__ pooled, int* __restrict__ counts, int N) {
  int nid = blockIdx.x * 8 + (threadIdx.x >> 5);
  int lane = threadIdx.x & 31;
  if (nid >= N) return;
  int g = batch[nid];
  float2 v = ((const float2*)(h + (size_t)nid * 64))[lane];
  atomicAdd(&pooled[(size_t)g * 64 + lane * 2],     v.x);
  atomicAdd(&pooled[(size_t)g * 64 + lane * 2 + 1], v.y);
  if (lane == 0) atomicAdd(&counts[g], 1);
}

__global__ void pool_div_kernel(float* pooled, const int* counts, int G) {
  int i = blockIdx.x * blockDim.x + threadIdx.x;
  if (i < G * 64) {
    int c = counts[i >> 6];
    pooled[i] /= (float)(c > 0 ? c : 1);
  }
}

// ---------------- host orchestration ----------------

extern "C" void kernel_launch(void* const* d_in, const int* in_sizes, int n_in,
                              void* d_out, int out_size, void* d_ws, size_t ws_size,
                              hipStream_t stream) {
  (void)n_in; (void)out_size; (void)ws_size;
  const float* x         = (const float*)d_in[0];
  const float* edge_attr = (const float*)d_in[1];
  const int*   edge_idx  = (const int*)d_in[2];
  const int*   batch     = (const int*)d_in[3];
  const float* W_dense   = (const float*)d_in[4];
  const float* fW1 = (const float*)d_in[5];
  const float* fb1 = (const float*)d_in[6];
  const float* fW2 = (const float*)d_in[7];
  const float* fb2 = (const float*)d_in[8];
  const float* attn_v = (const float*)d_in[9];
  const float* oW1 = (const float*)d_in[10];
  const float* ob1 = (const float*)d_in[11];
  const float* oW2 = (const float*)d_in[12];
  const float* ob2 = (const float*)d_in[13];
  const float* nW1 = (const float*)d_in[14];
  const float* nb1 = (const float*)d_in[15];
  const float* nW2 = (const float*)d_in[16];
  const float* nb2 = (const float*)d_in[17];

  const int N = in_sizes[0] / NODE_DIM;
  const int E = in_sizes[2] / 2;
  const int G = NUM_GRAPHS;
  const int* src = edge_idx;
  const int* tgt = edge_idx + E;

  float* ws = (float*)d_ws;
  float* h       = ws; ws += (size_t)N * 64;
  float* hd      = ws; ws += (size_t)N * 64;
  float* ew      = ws; ws += (size_t)E * 64;
  float* scores  = ws; ws += (size_t)E;
  float* denom   = ws; ws += (size_t)N;
  unsigned* segm = (unsigned*)ws; ws += (size_t)N;
  float* agg     = ws; ws += (size_t)N * 64;
  float* pooled  = ws; ws += (size_t)G * 64;
  int*   counts  = (int*)ws;

  hipMemcpyAsync(h, x, (size_t)N * 64 * sizeof(float), hipMemcpyDeviceToDevice, stream);

  dim3 blk(128);
  int nodeTiles        = (N + 63) / 64;
  int edgeTiles        = (E + 63) / 64;
  int edgeWarpBlocks   = (E + 7) / 8;
  int nodeWarpBlocks   = (N + 7) / 8;
  int edgeThreadBlocks = (E + 255) / 256;
  int nodeThreadBlocks = (N + 255) / 256;

  for (int l = 0; l < N_LAYERS; ++l) {
    dense64_kernel<<<nodeTiles, blk, 0, stream>>>(h, W_dense + (size_t)l * 64 * 64, hd, N);
    mlp2_kernel<EDGE_DIM, HIDDEN_D, NODE_DIM, false><<<edgeTiles, blk, 0, stream>>>(
        edge_attr, fW1 + (size_t)l * EDGE_DIM * HIDDEN_D, fb1 + (size_t)l * HIDDEN_D,
        fW2 + (size_t)l * HIDDEN_D * NODE_DIM, fb2 + (size_t)l * NODE_DIM,
        nullptr, ew, E);
    score_kernel<<<edgeWarpBlocks, 256, 0, stream>>>(hd, ew, src, attn_v + (size_t)l * 64,
                                                     scores, E);
    fill_u32_kernel<<<nodeThreadBlocks, 256, 0, stream>>>(segm, 0x007FFFFFu /*key(-inf)*/, N);
    segmax_kernel<<<edgeThreadBlocks, 256, 0, stream>>>(scores, tgt, segm, E);
    hipMemsetAsync(denom, 0, (size_t)N * sizeof(float), stream);
    expdenom_kernel<<<edgeThreadBlocks, 256, 0, stream>>>(scores, tgt, segm, denom, E);
    hipMemsetAsync(agg, 0, (size_t)N * 64 * sizeof(float), stream);
    scatter_kernel<<<edgeWarpBlocks, 256, 0, stream>>>(hd, ew, scores, denom, src, tgt, agg, E);
    mlp2_kernel<NODE_DIM, HIDDEN_D, NODE_DIM, true><<<nodeTiles, blk, 0, stream>>>(
        agg, oW1 + (size_t)l * NODE_DIM * HIDDEN_D, ob1 + (size_t)l * HIDDEN_D,
        oW2 + (size_t)l * HIDDEN_D * NODE_DIM, ob2 + (size_t)l * NODE_DIM,
        h, h, N);
  }

  hipMemsetAsync(pooled, 0, (size_t)G * 64 * sizeof(float), stream);
  hipMemsetAsync(counts, 0, (size_t)G * sizeof(int), stream);
  pool_accum_kernel<<<nodeWarpBlocks, 256, 0, stream>>>(h, batch, pooled, counts, N);
  pool_div_kernel<<<(G * 64 + 255) / 256, 256, 0, stream>>>(pooled, counts, G);
  mlp2_kernel<NODE_DIM, HIDDEN_D, OUT_DIM, false><<<(G + 63) / 64, blk, 0, stream>>>(
      pooled, nW1, nb1, nW2, nb2, nullptr, (float*)d_out, G);
}
